// QLiteMLA_9869834846699
// MI455X (gfx1250) — matrix operand order repaired and checked by reading, not compiled, not run on hardware
//
#include <hip/hip_runtime.h>

typedef __attribute__((ext_vector_type(16))) _Float16 v16h;
typedef __attribute__((ext_vector_type(8)))  _Float16 v8h;
typedef __attribute__((ext_vector_type(8)))  float    v8f;

#define NB    16          // batch
#define CIN   256         // input channels
#define NPIX  4096        // 64*64
#define C3T   768         // 3*TOTAL  (qkv channels)
#define CATT  512         // 2*TOTAL  (attention output channels)
#define GATT  64          // attention groups
#define ATT_EPS 1e-15f
#define BN_EPS  1e-5f

// ---------------------------------------------------------------------------
// WMMA 16x(16*NT) strip GEMM helper: one A fragment per K-step, NT B fragments
// and NT accumulators (independent WMMA chains for latency hiding).
// A0 : row base for this lane's M row of W (K-contiguous, f16)
// B0 : base of this lane's first N column (stored N-major, K-contiguous, f16);
//      column j is at B0 + j*16*K.
// k order used here (contiguous, same for A and B; not AMD's table order):
//   A halves 0..7  <- K = k0 + hi*16 .. +7     (VGPR0..3)
//   A halves 8..15 <- K = k0 + hi*16 + 8 ..   (VGPR4..7)
//   B halves 0..15 <- K = k0 + hi*16 .. +15   (N = lane&15)
// ---------------------------------------------------------------------------
template <int K, int NT>
__device__ __forceinline__ void gemm_strip_f16(const _Float16* __restrict__ A0,
                                               const _Float16* __restrict__ B0,
                                               int hi, v8f acc[NT])
{
#pragma unroll
    for (int kk = 0; kk < K; kk += 32) {
        const _Float16* ap = A0 + kk + hi * 16;
        v8h alo = *(const v8h*)(ap);
        v8h ahi = *(const v8h*)(ap + 8);
        v16h a = __builtin_shufflevector(alo, ahi,
                                         0, 1, 2, 3, 4, 5, 6, 7,
                                         8, 9, 10, 11, 12, 13, 14, 15);
#pragma unroll
        for (int j = 0; j < NT; ++j) {
            v16h b = *(const v16h*)(B0 + (size_t)j * 16 * K + kk + hi * 16);
            acc[j] = __builtin_amdgcn_wmma_f32_16x16x32_f16(
                /*neg_a=*/false, a, /*neg_b=*/false, b,
                /*c_mod=*/(short)0, acc[j], /*reuse_a=*/false, /*reuse_b=*/false);
        }
    }
}

// ---------------------------------------------------------------------------
// f32 -> f16 elementwise convert (weights)
// ---------------------------------------------------------------------------
__global__ void k_f32_to_f16(const float* __restrict__ in,
                             _Float16* __restrict__ out, int n)
{
    int i = blockIdx.x * blockDim.x + threadIdx.x;
    if (i < n) out[i] = (_Float16)in[i];
}

// ---------------------------------------------------------------------------
// x (B,C,N) f32  ->  xt (B,N,C) f16   via LDS 32x32 tile transpose
// grid: (NPIX/32, CIN/32, B), block: 256 (32x8)
// ---------------------------------------------------------------------------
__global__ void k_x_to_f16t(const float* __restrict__ x,
                            _Float16* __restrict__ xt)
{
    __shared__ float t[32][33];
    int tx = threadIdx.x & 31, ty = threadIdx.x >> 5;   // ty in 0..7
    int n0 = blockIdx.x * 32, c0 = blockIdx.y * 32, b = blockIdx.z;
    const float* xb = x + (size_t)b * CIN * NPIX;
#pragma unroll
    for (int i = 0; i < 4; ++i) {
        int c = c0 + ty + i * 8;
        t[ty + i * 8][tx] = xb[(size_t)c * NPIX + n0 + tx];
    }
    __syncthreads();
    _Float16* xtb = xt + (size_t)b * NPIX * CIN;
#pragma unroll
    for (int i = 0; i < 4; ++i) {
        int n = n0 + ty + i * 8;
        xtb[(size_t)n * CIN + c0 + tx] = (_Float16)t[tx][ty + i * 8];
    }
}

// ---------------------------------------------------------------------------
// qkv 1x1 conv as WMMA GEMM:  qkv[b] (768x4096) = Wqkv(768x256) @ X[b](256x4096)
// one wave per 16x64 output strip (4 accumulators, shared A fragment)
// waves = NB * 48 * 64 ; 8 waves per block
// ---------------------------------------------------------------------------
__global__ void k_gemm_qkv(const _Float16* __restrict__ Wh,
                           const _Float16* __restrict__ Xt,
                           float* __restrict__ qkv)
{
    int wid  = blockIdx.x * 8 + (threadIdx.x >> 5);
    int lane = threadIdx.x & 31;
    int l = lane & 15, hi = lane >> 4;
    int nt4 = wid % 64;                   // 64-column strip index
    int ot  = (wid / 64) % 48;
    int b   = wid / (64 * 48);
    int o   = ot * 16 + l;                // A-matrix row for this lane
    int n0  = nt4 * 64 + l;               // first B-matrix column for this lane
    v8f acc[4] = {};
    gemm_strip_f16<CIN, 4>(Wh + (size_t)o * CIN,
                           Xt + ((size_t)b * NPIX + n0) * CIN, hi, acc);
    float* outb = qkv + (size_t)b * C3T * NPIX;
#pragma unroll
    for (int j = 0; j < 4; ++j) {
        int col = nt4 * 64 + j * 16 + l;
#pragma unroll
        for (int r = 0; r < 8; ++r) {
            int row = ot * 16 + r + hi * 8;   // C/D layout: lanes16-31 hold M=8+r
            outb[(size_t)row * NPIX + col] = acc[j][r];
        }
    }
}

// ---------------------------------------------------------------------------
// depthwise 5x5 (pad 2) + grouped 8x8 pointwise, one 16x16 pixel tile per block
// grid = NB * 96 * 4 * 4, block = 256
// ---------------------------------------------------------------------------
__global__ void k_dwpw(const float* __restrict__ qkv,
                       const float* __restrict__ wdw,
                       const float* __restrict__ wpw,
                       float* __restrict__ agg)
{
    __shared__ float tile[8][20][20];
    __shared__ float wdwS[8][25];
    __shared__ float wpwS[8][8];
    int tid = threadIdx.x;
    int bi  = blockIdx.x;
    int twx = bi & 3, twy = (bi >> 2) & 3;
    int gp  = (bi >> 4) % 96;
    int b   = bi / (16 * 96);
    int h0 = twy * 16, w0 = twx * 16;
    const float* qb = qkv + ((size_t)b * C3T + gp * 8) * NPIX;
    if (tid < 200) wdwS[tid / 25][tid % 25] = wdw[(gp * 8 + tid / 25) * 25 + tid % 25];
    if (tid < 64)  wpwS[tid >> 3][tid & 7]  = wpw[(gp * 8 + (tid >> 3)) * 8 + (tid & 7)];
    for (int idx = tid; idx < 3200; idx += 256) {
        int cc = idx / 400, rem = idx % 400;
        int yy = rem / 20, xx = rem % 20;
        int h = h0 + yy - 2, w = w0 + xx - 2;
        float v = 0.f;
        if (h >= 0 && h < 64 && w >= 0 && w < 64)
            v = qb[(size_t)cc * NPIX + h * 64 + w];
        tile[cc][yy][xx] = v;
    }
    __syncthreads();
    int tx = tid & 15, ty = tid >> 4;
    float dwv[8];
#pragma unroll
    for (int cc = 0; cc < 8; ++cc) {
        float s = 0.f;
#pragma unroll
        for (int dy = 0; dy < 5; ++dy)
#pragma unroll
            for (int dx = 0; dx < 5; ++dx)
                s += tile[cc][ty + dy][tx + dx] * wdwS[cc][dy * 5 + dx];
        dwv[cc] = s;
    }
    float* ab = agg + ((size_t)b * C3T + gp * 8) * NPIX;
    int n = (h0 + ty) * 64 + (w0 + tx);
#pragma unroll
    for (int o = 0; o < 8; ++o) {
        float s = 0.f;
#pragma unroll
        for (int cc = 0; cc < 8; ++cc) s += wpwS[o][cc] * dwv[cc];
        ab[(size_t)o * NPIX + n] = s;
    }
}

// ---------------------------------------------------------------------------
// kv = sum_n relu(k) (x) [v,1] : one block per (b,g); shuffle+LDS reduction
// ---------------------------------------------------------------------------
__global__ void k_kv(const float* __restrict__ qkv,
                     const float* __restrict__ agg,
                     float* __restrict__ kvbuf)
{
    int bi = blockIdx.x;
    int g  = bi & 63, b = bi >> 6;
    const float* src = (g < 32)
        ? (qkv + ((size_t)b * C3T + 24 * g) * NPIX)
        : (agg + ((size_t)b * C3T + 24 * (g - 32)) * NPIX);
    float acc[72];
#pragma unroll
    for (int i = 0; i < 72; ++i) acc[i] = 0.f;
    for (int n = threadIdx.x; n < NPIX; n += 256) {
        float kv[8], vv[8];
#pragma unroll
        for (int d = 0; d < 8; ++d) {
            kv[d] = fmaxf(src[(size_t)(8 + d) * NPIX + n], 0.f);
            vv[d] = src[(size_t)(16 + d) * NPIX + n];
        }
#pragma unroll
        for (int d = 0; d < 8; ++d) {
#pragma unroll
            for (int e = 0; e < 8; ++e) acc[d * 9 + e] += kv[d] * vv[e];
            acc[d * 9 + 8] += kv[d];
        }
    }
    __shared__ float s_kv[72];
    if (threadIdx.x < 72) s_kv[threadIdx.x] = 0.f;
    __syncthreads();
    int lane = threadIdx.x & 31;
#pragma unroll
    for (int i = 0; i < 72; ++i) {
        float v = acc[i];
        for (int off = 16; off > 0; off >>= 1) v += __shfl_down(v, off, 32);
        if (lane == 0) atomicAdd(&s_kv[i], v);
    }
    __syncthreads();
    if (threadIdx.x < 72)
        kvbuf[(size_t)bi * 72 + threadIdx.x] = s_kv[threadIdx.x];
}

// ---------------------------------------------------------------------------
// out = relu(q) @ kv ; divide by denominator; emit f16, N-major [b][n][512]
// grid = NB * 64 * 16, block = 256 (one n per thread)
// ---------------------------------------------------------------------------
__global__ void k_att(const float* __restrict__ qkv,
                      const float* __restrict__ agg,
                      const float* __restrict__ kvbuf,
                      _Float16* __restrict__ att)
{
    int bi = blockIdx.x;
    int chunk = bi & 15, g = (bi >> 4) & 63, b = bi >> 10;
    __shared__ float s_kv[72];
    if (threadIdx.x < 72)
        s_kv[threadIdx.x] = kvbuf[((size_t)(b * 64 + g)) * 72 + threadIdx.x];
    __syncthreads();
    const float* src = (g < 32)
        ? (qkv + ((size_t)b * C3T + 24 * g) * NPIX)
        : (agg + ((size_t)b * C3T + 24 * (g - 32)) * NPIX);
    int n = chunk * 256 + threadIdx.x;
    float q[8];
#pragma unroll
    for (int d = 0; d < 8; ++d) q[d] = fmaxf(src[(size_t)d * NPIX + n], 0.f);
    float o[9];
#pragma unroll
    for (int e = 0; e < 9; ++e) {
        float s = 0.f;
#pragma unroll
        for (int d = 0; d < 8; ++d) s += q[d] * s_kv[d * 9 + e];
        o[e] = s;
    }
    float den = o[8] + ATT_EPS;
    v8h r;
#pragma unroll
    for (int d = 0; d < 8; ++d) r[d] = (_Float16)(o[d] / den);
    *(v8h*)(att + ((size_t)b * NPIX + n) * CATT + g * 8) = r;   // 16B aligned
}

// ---------------------------------------------------------------------------
// proj 1x1 conv WMMA GEMM + fused BatchNorm:
// out[b](256x4096) = Wproj(256x512) @ att[b](512x4096)
// one wave per 16x64 strip; waves = NB * 16 * 64
// ---------------------------------------------------------------------------
__global__ void k_gemm_proj_bn(const _Float16* __restrict__ Wh,
                               const _Float16* __restrict__ At,
                               const float* __restrict__ bng,
                               const float* __restrict__ bnb,
                               const float* __restrict__ bnm,
                               const float* __restrict__ bnv,
                               float* __restrict__ out)
{
    int wid  = blockIdx.x * 8 + (threadIdx.x >> 5);
    int lane = threadIdx.x & 31;
    int l = lane & 15, hi = lane >> 4;
    int nt4 = wid % 64;
    int ot  = (wid / 64) % 16;
    int b   = wid / (64 * 16);
    int o   = ot * 16 + l;
    int n0  = nt4 * 64 + l;
    v8f acc[4] = {};
    gemm_strip_f16<CATT, 4>(Wh + (size_t)o * CATT,
                            At + ((size_t)b * NPIX + n0) * CATT, hi, acc);
    float* ob = out + (size_t)b * CIN * NPIX;
#pragma unroll
    for (int r = 0; r < 8; ++r) {
        int row = ot * 16 + r + hi * 8;
        float inv = bng[row] * rsqrtf(bnv[row] + BN_EPS);
        float sh  = bnb[row] - bnm[row] * inv;
#pragma unroll
        for (int j = 0; j < 4; ++j) {
            int col = nt4 * 64 + j * 16 + l;
            ob[(size_t)row * NPIX + col] = acc[j][r] * inv + sh;
        }
    }
}

// ---------------------------------------------------------------------------
extern "C" void kernel_launch(void* const* d_in, const int* in_sizes, int n_in,
                              void* d_out, int out_size, void* d_ws, size_t ws_size,
                              hipStream_t stream)
{
    (void)in_sizes; (void)n_in; (void)out_size; (void)ws_size;
    const float* x      = (const float*)d_in[0];
    const float* w_qkv  = (const float*)d_in[1];   // 768*256
    const float* w_dw   = (const float*)d_in[2];   // 768*25
    const float* w_pw   = (const float*)d_in[3];   // 768*8
    const float* w_proj = (const float*)d_in[4];   // 256*512
    const float* bng    = (const float*)d_in[5];
    const float* bnb    = (const float*)d_in[6];
    const float* bnm    = (const float*)d_in[7];
    const float* bnv    = (const float*)d_in[8];
    float* out = (float*)d_out;

    // workspace carve-out (256B aligned slices)
    size_t off = 0;
    auto take = [&](size_t bytes) -> void* {
        void* p = (char*)d_ws + off;
        off += (bytes + 255) & ~(size_t)255;
        return p;
    };
    _Float16* Wqkv_h  = (_Float16*)take((size_t)C3T * CIN * 2);          // 384 KB
    _Float16* Wproj_h = (_Float16*)take((size_t)CIN * CATT * 2);         // 256 KB
    _Float16* xt      = (_Float16*)take((size_t)NB * NPIX * CIN * 2);    // 32 MB
    float*    qkv     = (float*)take((size_t)NB * C3T * NPIX * 4);       // 192 MB
    float*    agg     = (float*)take((size_t)NB * C3T * NPIX * 4);       // 192 MB
    float*    kvbuf   = (float*)take((size_t)NB * GATT * 72 * 4);        // 288 KB
    _Float16* att     = (_Float16*)take((size_t)NB * NPIX * CATT * 2);   // 64 MB

    // 1) weight conversion to f16
    k_f32_to_f16<<<(C3T * CIN + 255) / 256, 256, 0, stream>>>(w_qkv, Wqkv_h, C3T * CIN);
    k_f32_to_f16<<<(CIN * CATT + 255) / 256, 256, 0, stream>>>(w_proj, Wproj_h, CIN * CATT);
    // 2) x -> f16, N-major (B-fragment friendly)
    k_x_to_f16t<<<dim3(NPIX / 32, CIN / 32, NB), 256, 0, stream>>>(x, xt);
    // 3) qkv GEMM (WMMA, 16x64 strips)
    k_gemm_qkv<<<(NB * 48 * 64) / 8, 256, 0, stream>>>(Wqkv_h, xt, qkv);
    // 4) depthwise 5x5 + grouped 8x8 pointwise
    k_dwpw<<<NB * 96 * 16, 256, 0, stream>>>(qkv, w_dw, w_pw, agg);
    // 5) kv reduction per (b, g)
    k_kv<<<NB * GATT, 256, 0, stream>>>(qkv, agg, kvbuf);
    // 6) apply attention, emit f16 N-major
    k_att<<<NB * GATT * 16, 256, 0, stream>>>(qkv, agg, kvbuf, att);
    // 7) projection GEMM (WMMA, 16x64 strips) + fused BN -> output
    k_gemm_proj_bn<<<(NB * 16 * 64) / 8, 256, 0, stream>>>(
        Wproj_h, att, bng, bnb, bnm, bnv, out);
}
